// MambaBlock_77206332113651
// MI455X (gfx1250) — matrix operand order repaired and checked by reading, not compiled, not run on hardware
//
#include <hip/hip_runtime.h>
#include <hip/hip_bf16.h>
#include <stdint.h>

// ---------------- problem constants ----------------
#define D_MODEL 1024
#define D_INNER 2048
#define D_STATE 16
#define DT_RANK 64
#define XDBL_W  (DT_RANK + 2 * D_STATE)   // 96
#define BATCH   2
#define SEQ     1024
#define TOK     (BATCH * SEQ)             // 2048

// ---------------- CDNA5 vector types ----------------
typedef __attribute__((ext_vector_type(16))) __bf16 v16bf;
typedef __attribute__((ext_vector_type(8)))  float  v8f;
typedef __attribute__((ext_vector_type(8)))  unsigned short us8;

// ---------------- async global->LDS (CDNA5), with sync fallback ----------------
#if defined(__has_builtin)
#if __has_builtin(__builtin_amdgcn_global_load_async_to_lds_b128) && \
    __has_builtin(__builtin_amdgcn_s_wait_asynccnt)
#define USE_ASYNC_LDS 1
#endif
#endif
#ifndef USE_ASYNC_LDS
#define USE_ASYNC_LDS 0
#endif

typedef int v4i_vs __attribute__((vector_size(16)));          // int4 (GCC vector)
typedef __attribute__((address_space(1))) v4i_vs* gptr_v4i;   // global int4*
typedef __attribute__((address_space(3))) v4i_vs* lptr_v4i;   // LDS int4*

__device__ __forceinline__ void async_copy16(const void* g, void* l) {
#if USE_ASYNC_LDS
  __builtin_amdgcn_global_load_async_to_lds_b128((gptr_v4i)g, (lptr_v4i)l, 0, 0);
#else
  *(us8*)l = *(const us8*)g;
#endif
}
__device__ __forceinline__ void wait_async_lds() {
#if USE_ASYNC_LDS
  __builtin_amdgcn_s_wait_asynccnt(0);
#endif
}

__device__ __forceinline__ unsigned short f2bf(float f) {
  unsigned int u = __float_as_uint(f);
  u += 0x7FFFu + ((u >> 16) & 1u);       // round-to-nearest-even
  return (unsigned short)(u >> 16);
}

// ---------------- f32 -> bf16, transposed: out[N][K] = in[K][N] ----------------
__global__ __launch_bounds__(256) void cvtT_bf16_kernel(const float* __restrict__ in,
                                                        unsigned short* __restrict__ out,
                                                        int K, int N) {
  int idx = blockIdx.x * 256 + threadIdx.x;
  if (idx >= K * N) return;
  int n = idx / K, k = idx - n * K;
  out[idx] = f2bf(in[(size_t)k * N + n]);
}

// ---------------- LayerNorm -> bf16 ----------------
__global__ __launch_bounds__(256) void ln_bf16_kernel(const float* __restrict__ x,
                                                      const float* __restrict__ gamma,
                                                      const float* __restrict__ beta,
                                                      unsigned short* __restrict__ h) {
  const int row = blockIdx.x;            // 0..TOK-1
  const float* xr = x + (size_t)row * D_MODEL;
  __shared__ float red[256];
  float s = 0.f;
  for (int i = threadIdx.x; i < D_MODEL; i += 256) s += xr[i];
  red[threadIdx.x] = s; __syncthreads();
  for (int off = 128; off > 0; off >>= 1) {
    if (threadIdx.x < off) red[threadIdx.x] += red[threadIdx.x + off];
    __syncthreads();
  }
  const float mu = red[0] * (1.f / D_MODEL);
  __syncthreads();
  float v = 0.f;
  for (int i = threadIdx.x; i < D_MODEL; i += 256) { float d = xr[i] - mu; v += d * d; }
  red[threadIdx.x] = v; __syncthreads();
  for (int off = 128; off > 0; off >>= 1) {
    if (threadIdx.x < off) red[threadIdx.x] += red[threadIdx.x + off];
    __syncthreads();
  }
  const float rs = rsqrtf(red[0] * (1.f / D_MODEL) + 1e-5f);
  for (int i = threadIdx.x; i < D_MODEL; i += 256)
    h[(size_t)row * D_MODEL + i] = f2bf((xr[i] - mu) * rs * gamma[i] + beta[i]);
}

// ---------------- bf16 WMMA GEMM, double-buffered async LDS pipeline ----------
// C[M,N](f32) = A[M,K] @ Bt[N,K]^T (+ epilogue).
// A:  bf16 row-major [M][lda], K-contiguous.
// Bt: bf16 row-major [N][ldb] (pre-transposed weights), K-contiguous.
// Block tile 128x128, BK=32; 8 waves in 2x4 grid, each wave 64x32 = 4x2 WMMA
// tiles -> 8 v_wmma_f32_16x16x32_bf16 per K-step per wave.
#define BM 128
#define BN 128
#define BK 32
// epi: 0 = none; 1 = softplus(acc + aux[col]); 2 = acc + aux[row*ldc+col]
__global__ __launch_bounds__(256) void gemm_bf16_wmma_kernel(
    const unsigned short* __restrict__ A, int lda,
    const unsigned short* __restrict__ Bt, int ldb,
    float* __restrict__ C, int ldc,
    int M, int N, int K,
    int epi, const float* __restrict__ aux,
    unsigned short* __restrict__ Cbf) {
  __shared__ __align__(16) unsigned short sA[2][BM * BK];
  __shared__ __align__(16) unsigned short sB[2][BN * BK];

  const int tid  = threadIdx.x;
  const int lane = tid & 31;
  const int wave = tid >> 5;
  const int wr   = wave >> 2;       // 2 wave-rows  -> 64-row strips
  const int wc   = wave & 3;        // 4 wave-cols  -> 32-col strips
  const int rowBase = blockIdx.y * BM;
  const int colBase = blockIdx.x * BN;

  v8f zero = {};
  v8f acc[4][2];
#pragma unroll
  for (int m = 0; m < 4; ++m) { acc[m][0] = zero; acc[m][1] = zero; }

  // tile staging: 256 threads, 2 threads per 32-element row, 16B chunks
  const int tr  = tid >> 1;         // 0..127 : tile row (A: m-row, B: n-row)
  const int tc0 = (tid & 1) * 16;   // 0/16   : k-chunk within row

  const unsigned short* Arow = A + (size_t)(rowBase + tr) * lda + tc0;
  const int  bn     = colBase + tr;
  const bool bValid = bn < N;
  const unsigned short* Brow = Bt + (size_t)bn * ldb + tc0;

  auto stage = [&](int kt, int s) {
    unsigned short* da = &sA[s][tr * BK + tc0];
    async_copy16(Arow + kt, da);
    async_copy16(Arow + kt + 8, da + 8);
    unsigned short* db = &sB[s][tr * BK + tc0];
    if (bValid) {
      async_copy16(Brow + kt, db);
      async_copy16(Brow + kt + 8, db + 8);
    } else {
      us8 z = {};
      *(us8*)db = z; *(us8*)(db + 8) = z;
    }
  };

  const int nkt = K / BK;
  stage(0, 0);
  wait_async_lds();
  __syncthreads();

  const int arow  = wr * 64 + (lane & 15);
  const int bcol  = wc * 32 + (lane & 15);
  const int koffA = (lane >> 4) * 8;     // A frag: lanes 16-31 hold K=8..15/24..31
  const int koffB = (lane >> 4) * 8;     // B frag: lanes 16-31 hold K=8..15/24..31

  for (int it = 0; it < nkt; ++it) {
    const int cur = it & 1;
    if (it + 1 < nkt) stage((it + 1) * BK, cur ^ 1);   // overlap next tile

    union FragU { v16bf v; us8 h[2]; };
    FragU fa[4], fb[2];
#pragma unroll
    for (int m = 0; m < 4; ++m) {
      const unsigned short* p = &sA[cur][(arow + m * 16) * BK];
      fa[m].h[0] = *(const us8*)(p + koffA);
      fa[m].h[1] = *(const us8*)(p + 16 + koffA);
    }
#pragma unroll
    for (int n = 0; n < 2; ++n) {
      const unsigned short* p = &sB[cur][(bcol + n * 16) * BK + koffB];
      fb[n].h[0] = *(const us8*)p;
      fb[n].h[1] = *(const us8*)(p + 16);
    }
#pragma unroll
    for (int m = 0; m < 4; ++m)
#pragma unroll
      for (int n = 0; n < 2; ++n)
        acc[m][n] = __builtin_amdgcn_wmma_f32_16x16x32_bf16(
            false, fa[m].v, false, fb[n].v, (short)0, acc[m][n], false, false);

    wait_async_lds();   // own async writes to next stage complete
    __syncthreads();    // everyone done reading cur / writing next
  }

  // ---- epilogue: C layout VGPR j -> M=j (lanes 0-15) / M=j+8 (lanes 16-31) ----
  const int rhalf = (lane >> 4) * 8;
  const int cIn   = lane & 15;
#pragma unroll
  for (int m = 0; m < 4; ++m)
#pragma unroll
    for (int n = 0; n < 2; ++n) {
      const int trow = rowBase + wr * 64 + m * 16 + rhalf;
      const int tcol = colBase + wc * 32 + n * 16 + cIn;
      if (tcol < N) {
#pragma unroll
        for (int e = 0; e < 8; ++e) {
          const int row = trow + e;
          float v = acc[m][n][e];
          if (epi == 1) { v += aux[tcol]; v = (v > 20.f) ? v : log1pf(__expf(v)); }
          else if (epi == 2) { v += aux[(size_t)row * ldc + tcol]; }
          C[(size_t)row * ldc + tcol] = v;
          if (Cbf) Cbf[(size_t)row * ldc + tcol] = f2bf(v);
        }
      }
    }
}

// ---------------- causal depthwise conv(4) + silu -> f32 + bf16 ----------------
__global__ __launch_bounds__(256) void conv_silu_kernel(const float* __restrict__ xz,
                                                        const float* __restrict__ Wc,
                                                        const float* __restrict__ bc,
                                                        float* __restrict__ u,
                                                        unsigned short* __restrict__ ubf) {
  const int idx = blockIdx.x * 256 + threadIdx.x;   // over TOK*D_INNER
  const int d = idx & (D_INNER - 1);
  const int t = (idx >> 11) & (SEQ - 1);
  const int b = idx >> 21;
  const float* base = xz + ((size_t)b * SEQ) * (2 * D_INNER) + d;  // xi slice
  float a = bc[d];
#pragma unroll
  for (int j = 0; j < 4; ++j) {
    const int tt = t - 3 + j;
    if (tt >= 0) a += base[(size_t)tt * (2 * D_INNER)] * Wc[d * 4 + j];
  }
  const float s = a / (1.f + __expf(-a));
  u[idx]   = s;
  ubf[idx] = f2bf(s);
}

// ---------------- selective scan + Dskip + silu(z) gate -> bf16 ----------------
__global__ __launch_bounds__(256) void scan_kernel(const float* __restrict__ xz,
                                                   const float* __restrict__ xdbl,
                                                   const float* __restrict__ dt,
                                                   const float* __restrict__ u,
                                                   const float* __restrict__ A_log,
                                                   const float* __restrict__ Dskip,
                                                   unsigned short* __restrict__ ybf) {
  const int b = blockIdx.y;
  const int d = blockIdx.x * 256 + threadIdx.x;     // channel 0..D_INNER-1
  const float LOG2E = 1.4426950408889634f;
  float A2[D_STATE], h[D_STATE];
#pragma unroll
  for (int s = 0; s < D_STATE; ++s) {
    A2[s] = -__expf(A_log[d * D_STATE + s]) * LOG2E;   // A * log2(e)
    h[s] = 0.f;
  }
  const float Dv = Dskip[d];
  __shared__ float sB[32][D_STATE];
  __shared__ float sC[32][D_STATE];
  const size_t base = (size_t)b * SEQ;
  for (int t0 = 0; t0 < SEQ; t0 += 32) {
    __syncthreads();
#pragma unroll
    for (int i = 0; i < 4; ++i) {                     // 32 timesteps x 32 vals
      const int e = threadIdx.x + i * 256;
      const int tt = e >> 5, s = e & 31;
      const float v = xdbl[(base + t0 + tt) * XDBL_W + DT_RANK + s];
      if (s < D_STATE) sB[tt][s] = v; else sC[tt][s - D_STATE] = v;
    }
    __syncthreads();
    for (int tt = 0; tt < 32; ++tt) {
      const size_t off = (base + t0 + tt) * D_INNER + d;
      const float dtv = dt[off];
      const float uv  = u[off];
      const float du  = dtv * uv;
      float y = 0.f;
#pragma unroll
      for (int s = 0; s < D_STATE; ++s) {
        const float dA = exp2f(dtv * A2[s]);
        h[s] = dA * h[s] + du * sB[tt][s];
        y += h[s] * sC[tt][s];
      }
      const float zv = xz[(base + t0 + tt) * (2 * D_INNER) + D_INNER + d];
      const float g = zv / (1.f + __expf(-zv));       // silu(z)
      ybf[off] = f2bf((y + uv * Dv) * g);
    }
  }
}

// ---------------- launch ----------------
extern "C" void kernel_launch(void* const* d_in, const int* in_sizes, int n_in,
                              void* d_out, int out_size, void* d_ws, size_t ws_size,
                              hipStream_t stream) {
  (void)in_sizes; (void)n_in; (void)out_size; (void)ws_size;
  const float* x      = (const float*)d_in[0];
  const float* gamma  = (const float*)d_in[1];
  const float* beta   = (const float*)d_in[2];
  const float* W_in   = (const float*)d_in[3];
  const float* W_conv = (const float*)d_in[4];
  const float* b_conv = (const float*)d_in[5];
  const float* W_x    = (const float*)d_in[6];
  const float* W_dt   = (const float*)d_in[7];
  const float* b_dt   = (const float*)d_in[8];
  const float* A_log  = (const float*)d_in[9];
  const float* Dskip  = (const float*)d_in[10];
  const float* W_out  = (const float*)d_in[11];
  float* out = (float*)d_out;

  char* ws = (char*)d_ws;
  auto carve = [&](size_t bytes) -> char* {
    char* p = ws; ws += (bytes + 255) & ~(size_t)255; return p;
  };
  unsigned short* h_bf    = (unsigned short*)carve((size_t)TOK * D_MODEL * 2);
  unsigned short* WinT    = (unsigned short*)carve((size_t)D_MODEL * 2 * D_INNER * 2);
  unsigned short* WxT     = (unsigned short*)carve((size_t)D_INNER * XDBL_W * 2);
  unsigned short* WdtT    = (unsigned short*)carve((size_t)DT_RANK * D_INNER * 2);
  unsigned short* WoutT   = (unsigned short*)carve((size_t)D_INNER * D_MODEL * 2);
  float*          xz      = (float*)carve((size_t)TOK * 2 * D_INNER * 4);
  float*          u       = (float*)carve((size_t)TOK * D_INNER * 4);
  unsigned short* u_bf    = (unsigned short*)carve((size_t)TOK * D_INNER * 2);
  float*          xdbl    = (float*)carve((size_t)TOK * XDBL_W * 4);
  unsigned short* xdbl_bf = (unsigned short*)carve((size_t)TOK * XDBL_W * 2);
  float*          dtbuf   = (float*)carve((size_t)TOK * D_INNER * 4);
  unsigned short* y_bf    = (unsigned short*)carve((size_t)TOK * D_INNER * 2);

  // 1) weights -> bf16, transposed to [N][K] so GEMM B-tiles stage contiguously
  auto cvtT = [&](const float* src, unsigned short* dst, int K, int N) {
    int n = K * N;
    cvtT_bf16_kernel<<<(n + 255) / 256, 256, 0, stream>>>(src, dst, K, N);
  };
  cvtT(W_in,  WinT,  D_MODEL, 2 * D_INNER);
  cvtT(W_x,   WxT,   D_INNER, XDBL_W);
  cvtT(W_dt,  WdtT,  DT_RANK, D_INNER);
  cvtT(W_out, WoutT, D_INNER, D_MODEL);

  // 2) layernorm -> bf16 h
  ln_bf16_kernel<<<TOK, 256, 0, stream>>>(x, gamma, beta, h_bf);

  // 3) xz = h @ W_in                    (2048 x 4096, K=1024)
  gemm_bf16_wmma_kernel<<<dim3((2 * D_INNER) / BN, TOK / BM), 256, 0, stream>>>(
      h_bf, D_MODEL, WinT, D_MODEL, xz, 2 * D_INNER,
      TOK, 2 * D_INNER, D_MODEL, 0, nullptr, nullptr);

  // 4) u = silu(conv1d(xi) + b_conv)    (f32 for scan, bf16 for next GEMM)
  conv_silu_kernel<<<(TOK * D_INNER) / 256, 256, 0, stream>>>(xz, W_conv, b_conv, u, u_bf);

  // 5) x_dbl = u @ W_x                  (2048 x 96, K=2048) ; bf16 copy for dt GEMM
  gemm_bf16_wmma_kernel<<<dim3((XDBL_W + BN - 1) / BN, TOK / BM), 256, 0, stream>>>(
      u_bf, D_INNER, WxT, D_INNER, xdbl, XDBL_W,
      TOK, XDBL_W, D_INNER, 0, nullptr, xdbl_bf);

  // 6) dt = softplus(dt_r @ W_dt + b_dt)   (2048 x 2048, K=64; A = x_dbl[:, :64])
  gemm_bf16_wmma_kernel<<<dim3(D_INNER / BN, TOK / BM), 256, 0, stream>>>(
      xdbl_bf, XDBL_W, WdtT, DT_RANK, dtbuf, D_INNER,
      TOK, D_INNER, DT_RANK, 1, b_dt, nullptr);

  // 7) selective scan fused with Dskip + silu(z) gate -> y (bf16)
  scan_kernel<<<dim3(D_INNER / 256, BATCH), 256, 0, stream>>>(
      xz, xdbl, dtbuf, u, A_log, Dskip, y_bf);

  // 8) out = x + y @ W_out              (2048 x 1024, K=2048)
  gemm_bf16_wmma_kernel<<<dim3(D_MODEL / BN, TOK / BM), 256, 0, stream>>>(
      y_bf, D_INNER, WoutT, D_INNER, out, D_MODEL,
      TOK, D_MODEL, D_INNER, 2, x, nullptr);
}
